// Block_82592221102594
// MI455X (gfx1250) — compile-verified, matrix operand order checked by reading, not run on hardware
//
#include <hip/hip_runtime.h>
#include <math.h>

static constexpr int C_L = 1024, C_B = 8, C_E = 512, C_H = 8, C_HD = 64, C_NE = 8;
static constexpr int C_TOK = C_L * C_B;                 // 8192 tokens
static constexpr size_t C_TE = (size_t)C_TOK * C_E;     // 4,194,304 floats per token-matrix

typedef __attribute__((ext_vector_type(16))) __bf16 v16bf;
typedef __attribute__((ext_vector_type(8)))  float  v8f;

__device__ __forceinline__ float2 ldg2(const float* __restrict__ p, unsigned off) {
    return *(const float2*)(p + (size_t)off);   // uniform base + u32 offset -> saddr form
}

// ---------------------------------------------------------------------------
// WMMA GEMM:  C[m,n] (+)= comb[m] * act( alpha * sum_k A[m,k]*W'[n,k] + bias[n] )
//   WTRANS==0: W is (N,K) row-major.  WTRANS==1: W is (K,N) row-major.
//   grid.z adds per-slice pointer offsets (attention heads).
//   Block: 128 threads = 4 waves. Block tile 128M x 64N; each wave owns
//   32M x 64N (2x4 accumulators -> 8 WMMAs per K-chunk, A reused across N).
//   B K-chunk (32K x 64N) is staged in LDS as bf16, K-contiguous with a
//   40-element row stride so each fragment is two 16B-aligned ds_load_b128.
//   M%128==0, N%64==0, Kdim%32==0. EXEC all-ones throughout.
// ---------------------------------------------------------------------------
template<int WTRANS>
__global__ __launch_bounds__(128)
void wmma_gemm(const float* __restrict__ A, int lda, int sAz,
               const float* __restrict__ W, int ldw, int sWz,
               const float* __restrict__ bias, float alpha, int act,
               const float* __restrict__ comb, int comb_stride, int accum,
               float* __restrict__ C, int ldc, int sCz, int Kdim)
{
    __shared__ __bf16 Bs[64][40];   // [n][k], k-padded: 80B rows (16B aligned)

    const float* Ab = A + (size_t)blockIdx.z * (size_t)sAz;
    const float* Wb = W + (size_t)blockIdx.z * (size_t)sWz;
    float*       Cb = C + (size_t)blockIdx.z * (size_t)sCz;

    const int tid  = threadIdx.x;
    const int lane = tid & 31;
    const int wave = tid >> 5;
    const int g    = lane >> 4;          // K-half select per ISA fragment layout
    const int ln16 = lane & 15;
    const int m_base = blockIdx.y * 128 + wave * 32;
    const int n_base = blockIdx.x * 64;

    // A row offsets (u32, uniform base Ab): two 16-row tiles per wave.
    const unsigned offA0 = (unsigned)((m_base + ln16) * lda) + 8u * (unsigned)g;
    const unsigned offA1 = offA0 + (unsigned)(16 * lda);

    v8f acc[2][4] = {};

    for (int kb = 0; kb < Kdim; kb += 32) {
        // ---- cooperative fill of Bs[n][k] = W'[n_base+n][kb+k] (bf16) ----
        if (WTRANS == 0) {
            // W (N,K): read along k (coalesced b64), pack pair into Bs row.
            const int n0 = tid >> 4;            // 0..7
            const int k0 = (tid & 15) * 2;      // 0..30 even
#pragma unroll
            for (int r = 0; r < 8; ++r) {
                const int n = n0 + r * 8;
                const float2 w = ldg2(Wb, (unsigned)((n_base + n) * ldw + kb + k0));
                Bs[n][k0]     = (__bf16)w.x;
                Bs[n][k0 + 1] = (__bf16)w.y;
            }
        } else {
            // W (K,N): read along n (coalesced b64), scatter to two Bs rows.
            const int k0 = tid >> 5;            // 0..3
            const int c0 = (tid & 31) * 2;      // 0..62 even
#pragma unroll
            for (int r = 0; r < 8; ++r) {
                const int k = k0 + r * 4;
                const float2 w = ldg2(Wb, (unsigned)((kb + k) * ldw + n_base + c0));
                Bs[c0][k]     = (__bf16)w.x;
                Bs[c0 + 1][k] = (__bf16)w.y;
            }
        }
        __syncthreads();

        // ---- A fragments: direct global b64 loads, imm offsets off u32 base ----
        v16bf a0, a1;
#pragma unroll
        for (int v = 0; v < 8; ++v) {
            const int off = kb + ((v < 4) ? 2 * v : 8 + 2 * v);
            const float2 x0 = ldg2(Ab, offA0 + (unsigned)off);
            const float2 x1 = ldg2(Ab, offA1 + (unsigned)off);
            a0[2 * v]     = (__bf16)x0.x;  a0[2 * v + 1] = (__bf16)x0.y;
            a1[2 * v]     = (__bf16)x1.x;  a1[2 * v + 1] = (__bf16)x1.y;
        }

        // ---- B fragments from LDS (2x ds_load_b128 each) + 8 WMMAs ----
#pragma unroll
        for (int nt = 0; nt < 4; ++nt) {
            const __bf16* brow = &Bs[nt * 16 + ln16][8 * g];
            const uint4 lo = *(const uint4*)(brow);        // K = 8g .. 8g+7
            const uint4 hi = *(const uint4*)(brow + 16);   // K = 8g+16 .. 8g+23
            v16bf b;
            ((uint4*)&b)[0] = lo;
            ((uint4*)&b)[1] = hi;
            acc[0][nt] = __builtin_amdgcn_wmma_f32_16x16x32_bf16(false, a0, false, b,
                                                                 (short)0, acc[0][nt],
                                                                 false, false);
            acc[1][nt] = __builtin_amdgcn_wmma_f32_16x16x32_bf16(false, a1, false, b,
                                                                 (short)0, acc[1][nt],
                                                                 false, false);
        }
        __syncthreads();
    }

    // ---- epilogue ----
#pragma unroll
    for (int mt = 0; mt < 2; ++mt) {
#pragma unroll
        for (int nt = 0; nt < 4; ++nt) {
            const int ncol = n_base + nt * 16 + ln16;
            const float bval = bias ? bias[ncol] : 0.f;
#pragma unroll
            for (int r = 0; r < 8; ++r) {
                const int m = m_base + mt * 16 + r + 8 * g;  // C/D layout
                float val = alpha * acc[mt][nt][r] + bval;
                if (act)  val = (val >= 0.f) ? val : 0.01f * val;  // LeakyReLU
                if (comb) val *= comb[(size_t)m * comb_stride];
                const unsigned idx = (unsigned)(m * ldc + ncol);
                if (accum) Cb[idx] += val; else Cb[idx] = val;
            }
        }
    }
}

// ---------------------------------------------------------------------------
// Wave + LDS reductions (wave32)
// ---------------------------------------------------------------------------
__device__ inline float wred_sum(float v) {
#pragma unroll
    for (int o = 16; o > 0; o >>= 1) v += __shfl_down(v, o);
    return v;
}
__device__ inline float wred_max(float v) {
#pragma unroll
    for (int o = 16; o > 0; o >>= 1) v = fmaxf(v, __shfl_down(v, o));
    return v;
}

// One block (256 thr) per score row; grid = (L, H). In-place softmax.
__global__ __launch_bounds__(256)
void softmax_rows(float* __restrict__ sc, int Lr)
{
    float* row = sc + ((size_t)blockIdx.y * Lr + blockIdx.x) * (size_t)Lr;
    __shared__ float sm[8];
    const int tid = threadIdx.x;

    float mx = -INFINITY;
    for (int i = tid; i < Lr; i += 256) mx = fmaxf(mx, row[i]);
    mx = wred_max(mx);
    if ((tid & 31) == 0) sm[tid >> 5] = mx;
    __syncthreads();
    float m2 = sm[0];
#pragma unroll
    for (int w = 1; w < 8; ++w) m2 = fmaxf(m2, sm[w]);
    __syncthreads();

    float s = 0.f;
    for (int i = tid; i < Lr; i += 256) {
        const float e = expf(row[i] - m2);
        row[i] = e; s += e;
    }
    s = wred_sum(s);
    if ((tid & 31) == 0) sm[tid >> 5] = s;
    __syncthreads();
    float tot = 0.f;
#pragma unroll
    for (int w = 0; w < 8; ++w) tot += sm[w];
    const float inv = 1.f / tot;
    for (int i = tid; i < Lr; i += 256) row[i] *= inv;
}

// out = LayerNorm(a (+ b) (+ c)) * gamma + beta ; one block per token row.
__global__ __launch_bounds__(256)
void add_ln(const float* __restrict__ a, const float* __restrict__ b,
            const float* __restrict__ c, const float* __restrict__ gamma,
            const float* __restrict__ beta, float* __restrict__ out, int E)
{
    const size_t base = (size_t)blockIdx.x * E;
    __shared__ float s1s[8], s2s[8];
    const int tid = threadIdx.x;

    float s = 0.f, s2 = 0.f;
    for (int i = tid; i < E; i += 256) {
        float v = a[base + i];
        if (b) v += b[base + i];
        if (c) v += c[base + i];
        s += v; s2 += v * v;
    }
    s = wred_sum(s); s2 = wred_sum(s2);
    if ((tid & 31) == 0) { s1s[tid >> 5] = s; s2s[tid >> 5] = s2; }
    __syncthreads();
    float S = 0.f, S2 = 0.f;
#pragma unroll
    for (int w = 0; w < 8; ++w) { S += s1s[w]; S2 += s2s[w]; }
    const float mean = S / E;
    const float var  = S2 / E - mean * mean;
    const float inv  = rsqrtf(var + 1e-5f);
    for (int i = tid; i < E; i += 256) {
        float v = a[base + i];
        if (b) v += b[base + i];
        if (c) v += c[base + i];
        out[base + i] = (v - mean) * inv * gamma[i] + beta[i];
    }
}

// Per-token gating: softmax over 8 experts, top-2, normalized combine weights
// into wts (TOK x NE, zeros elsewhere); accumulate f and P sums for aux loss.
__global__ __launch_bounds__(128)
void gate_kernel(const float* __restrict__ x1, const float* __restrict__ gw,
                 const float* __restrict__ gb, float* __restrict__ wts,
                 float* __restrict__ fsum, float* __restrict__ psum, int ntok)
{
    const int t = blockIdx.x * blockDim.x + threadIdx.x;
    if (t >= ntok) return;
    float logit[C_NE];
#pragma unroll
    for (int e = 0; e < C_NE; ++e) logit[e] = gb[e];
    const float* xr = x1 + (size_t)t * C_E;
    for (int k = 0; k < C_E; k += 4) {
        const float4 xv = *(const float4*)(xr + k);
#pragma unroll
        for (int e = 0; e < C_NE; ++e) {
            const float4 wv = *(const float4*)(gw + e * C_E + k);
            logit[e] += xv.x * wv.x + xv.y * wv.y + xv.z * wv.z + xv.w * wv.w;
        }
    }
    float mx = logit[0];
#pragma unroll
    for (int e = 1; e < C_NE; ++e) mx = fmaxf(mx, logit[e]);
    float p[C_NE]; float s = 0.f;
#pragma unroll
    for (int e = 0; e < C_NE; ++e) { p[e] = expf(logit[e] - mx); s += p[e]; }
    const float invs = 1.f / s;
#pragma unroll
    for (int e = 0; e < C_NE; ++e) p[e] *= invs;

    int i0 = 0;
#pragma unroll
    for (int e = 1; e < C_NE; ++e) if (p[e] > p[i0]) i0 = e;
    int i1 = (i0 == 0) ? 1 : 0;
#pragma unroll
    for (int e = 0; e < C_NE; ++e) if (e != i0 && p[e] > p[i1]) i1 = e;

    const float wsum = p[i0] + p[i1];
    float* wr = wts + (size_t)t * C_NE;
#pragma unroll
    for (int e = 0; e < C_NE; ++e) wr[e] = 0.f;
    wr[i0] = p[i0] / wsum;
    wr[i1] = p[i1] / wsum;

    atomicAdd(&fsum[i0], 1.f);
    atomicAdd(&fsum[i1], 1.f);
#pragma unroll
    for (int e = 0; e < C_NE; ++e) atomicAdd(&psum[e], p[e]);
}

__global__ void zero_kernel(float* __restrict__ p, int n)
{
    for (int i = blockIdx.x * blockDim.x + threadIdx.x; i < n;
         i += gridDim.x * blockDim.x) p[i] = 0.f;
}

__global__ void loss_kernel(const float* __restrict__ fs, const float* __restrict__ ps,
                            float* __restrict__ out, float invM)
{
    if (threadIdx.x == 0 && blockIdx.x == 0) {
        float acc = 0.f;
#pragma unroll
        for (int e = 0; e < C_NE; ++e) acc += (fs[e] * invM) * (ps[e] * invM);
        out[0] = (float)C_NE * acc;
    }
}

// ---------------------------------------------------------------------------
extern "C" void kernel_launch(void* const* d_in, const int* in_sizes, int n_in,
                              void* d_out, int out_size, void* d_ws, size_t ws_size,
                              hipStream_t stream)
{
    (void)in_sizes; (void)n_in; (void)out_size; (void)ws_size;
    const float* x     = (const float*)d_in[0];
    const float* wq    = (const float*)d_in[1];
    const float* bq    = (const float*)d_in[2];
    const float* wk    = (const float*)d_in[3];
    const float* bk    = (const float*)d_in[4];
    const float* wv    = (const float*)d_in[5];
    const float* bv    = (const float*)d_in[6];
    const float* in_w  = (const float*)d_in[7];
    const float* in_b  = (const float*)d_in[8];
    const float* out_w = (const float*)d_in[9];
    const float* out_b = (const float*)d_in[10];
    const float* ln1g  = (const float*)d_in[11];
    const float* ln1b  = (const float*)d_in[12];
    const float* ln2g  = (const float*)d_in[13];
    const float* ln2b  = (const float*)d_in[14];
    const float* shw   = (const float*)d_in[15];
    const float* shb   = (const float*)d_in[16];
    const float* gw    = (const float*)d_in[17];
    const float* gb    = (const float*)d_in[18];
    const float* expw  = (const float*)d_in[19];
    const float* expb  = (const float*)d_in[20];

    // Workspace layout (floats): 4 token-slots + 2-slot score region + 1 slot + small
    float* ws  = (float*)d_ws;
    float* S0  = ws;                      // scratch / x1
    float* S1  = ws + 1 * C_TE;           // q1 / ao_proj
    float* S2  = ws + 2 * C_TE;           // k1 / moe_acc
    float* S3  = ws + 3 * C_TE;           // v1 / shared
    float* SC  = ws + 4 * C_TE;           // attention scores: H*L*L = 2*C_TE floats
    float* S5  = ws + 6 * C_TE;           // attention output (pre out_proj)
    float* WTS = ws + 7 * C_TE;           // TOK*NE combine weights
    float* FS  = WTS + (size_t)C_TOK * C_NE;  // NE expert-count sums
    float* PS  = FS + C_NE;                   // NE prob sums

    auto gemm = [&](const float* A, int lda, int sAz,
                    const float* W, int ldw, int sWz, int wtrans,
                    const float* bias, float alpha, int act,
                    const float* comb, int cstride, int accum,
                    float* C, int ldc, int sCz,
                    int M, int N, int Kd, int Z) {
        dim3 grid(N / 64, M / 128, Z), block(128);
        if (wtrans)
            wmma_gemm<1><<<grid, block, 0, stream>>>(A, lda, sAz, W, ldw, sWz,
                bias, alpha, act, comb, cstride, accum, C, ldc, sCz, Kd);
        else
            wmma_gemm<0><<<grid, block, 0, stream>>>(A, lda, sAz, W, ldw, sWz,
                bias, alpha, act, comb, cstride, accum, C, ldc, sCz, Kd);
    };

    const int E = C_E, T = C_TOK, BE = C_B * C_E;

    // ---- outer QKV + in_proj (q0->q1 etc.; S0 reused as staging) ----
    gemm(x,  E, 0, wq, E, 0, 0, bq, 1.f, 0, nullptr, 0, 0, S0, E, 0, T, E, E, 1);
    gemm(S0, E, 0, in_w,                     E, 0, 0, in_b,          1.f, 0, nullptr, 0, 0, S1, E, 0, T, E, E, 1);
    gemm(x,  E, 0, wk, E, 0, 0, bk, 1.f, 0, nullptr, 0, 0, S0, E, 0, T, E, E, 1);
    gemm(S0, E, 0, in_w + (size_t)E * E,     E, 0, 0, in_b + E,      1.f, 0, nullptr, 0, 0, S2, E, 0, T, E, E, 1);
    gemm(x,  E, 0, wv, E, 0, 0, bv, 1.f, 0, nullptr, 0, 0, S0, E, 0, T, E, E, 1);
    gemm(S0, E, 0, in_w + 2 * (size_t)E * E, E, 0, 0, in_b + 2 * E,  1.f, 0, nullptr, 0, 0, S3, E, 0, T, E, E, 1);

    // ---- attention, one batch at a time (scores scratch = 32 MB) ----
    for (int b = 0; b < C_B; ++b) {
        // scores[h,q,k] = (1/8) * q1[q] . k1[k]
        gemm(S1 + (size_t)b * E, BE, C_HD,
             S2 + (size_t)b * E, BE, C_HD, 0,
             nullptr, 0.125f, 0, nullptr, 0, 0,
             SC, C_L, C_L * C_L,
             C_L, C_L, C_HD, C_H);
        softmax_rows<<<dim3(C_L, C_H), 256, 0, stream>>>(SC, C_L);
        // ao[q,d] = sum_k att[q,k] * v1[k,d]  (B not transposed -> wtrans=1)
        gemm(SC, C_L, C_L * C_L,
             S3 + (size_t)b * E, BE, C_HD, 1,
             nullptr, 1.f, 0, nullptr, 0, 0,
             S5 + (size_t)b * E, BE, C_HD,
             C_L, C_HD, C_L, C_H);
    }

    // ---- out_proj, residual + LN1 ----
    gemm(S5, E, 0, out_w, E, 0, 0, out_b, 1.f, 0, nullptr, 0, 0, S1, E, 0, T, E, E, 1);
    add_ln<<<T, 256, 0, stream>>>(x, S1, nullptr, ln1g, ln1b, S0, E);   // x1 -> S0

    // ---- gating + MoE (fused combine-accumulate, no allout tensor) ----
    zero_kernel<<<1024, 256, 0, stream>>>(S2, (int)C_TE);               // moe_acc -> S2
    zero_kernel<<<1, 32, 0, stream>>>(FS, 2 * C_NE);
    gate_kernel<<<T / 128, 128, 0, stream>>>(S0, gw, gb, WTS, FS, PS, T);
    for (int n = 0; n < C_NE; ++n) {
        gemm(S0, E, 0, expw + (size_t)n * E * E, E, 0, 0,
             expb + (size_t)n * E, 1.f, /*leaky=*/1,
             WTS + n, C_NE, /*accum=*/1,
             S2, E, 0, T, E, E, 1);
    }

    // ---- shared expert, final residual + LN2, aux loss ----
    gemm(S0, E, 0, shw, E, 0, 0, shb, 1.f, 1, nullptr, 0, 0, S3, E, 0, T, E, E, 1);
    add_ln<<<T, 256, 0, stream>>>(S0, S2, S3, ln2g, ln2b, (float*)d_out, E);
    loss_kernel<<<1, 1, 0, stream>>>(FS, PS, (float*)d_out + C_TE, 1.f / (float)T);
}
